// TransformerBlock_39805756899526
// MI455X (gfx1250) — matrix operand order repaired and checked by reading, not compiled, not run on hardware
//
#include <hip/hip_runtime.h>
#include <hip/hip_bf16.h>
#include <math.h>
#include <stdint.h>

// ---------------------------------------------------------------------------
// CDNA5 (gfx1250) wave32 WMMA types
// ---------------------------------------------------------------------------
typedef __attribute__((ext_vector_type(16))) _Float16 v16h;
typedef __attribute__((ext_vector_type(8)))  float    v8f;

union AF { v16h v; _Float16 e[16]; };
union CF { v8f  v; float    e[8];  };

static __device__ __forceinline__ v8f wmma_f16(v16h a, v16h b, v8f c) {
    // D = A(16x32 f16) * B(32x16 f16) + C(16x16 f32)
    return __builtin_amdgcn_wmma_f32_16x16x32_f16(
        /*neg_a=*/false, a, /*neg_b=*/false, b,
        /*c_mod=*/(short)0, c, /*reuse_a=*/false, /*reuse_b=*/false);
}

// ---------------------------------------------------------------------------
// CDNA5 async memory->LDS copy (ASYNCcnt-tracked, VGPR-free data path).
// lds: byte address inside the LDS window (low 32 bits of a generic pointer
// to __shared__). gaddr: 64-bit global address.
// ---------------------------------------------------------------------------
static __device__ __forceinline__ void async_b128(uint32_t lds, uint64_t gaddr) {
    asm volatile("global_load_async_to_lds_b128 %0, %1, off"
                 :: "v"(lds), "v"(gaddr)
                 : "memory");
}
static __device__ __forceinline__ void wait_async0() {
    asm volatile("s_wait_asynccnt 0" ::: "memory");
}

#define B_    4
#define L_    4096
#define DIM_  512
#define H_    8
#define HD_   64
#define CHUNK_ 128

// ---------------------------------------------------------------------------
// LayerNorm (+optional rel_bias) -> f16 staging
// one block (256 thr) per token row of 512
// ---------------------------------------------------------------------------
__global__ void __launch_bounds__(256)
ln_kernel(const float* __restrict__ x, const float* __restrict__ w,
          const float* __restrict__ b, const float* __restrict__ rel,
          _Float16* __restrict__ out, int hasRel)
{
    __shared__ float rsum[8], rsq[8];
    const int row = blockIdx.x, t = threadIdx.x;
    const int wave = t >> 5, lane = t & 31;
    const float* xr = x + (size_t)row * DIM_;
    float v0 = xr[t], v1 = xr[t + 256];
    float s1 = v0 + v1, s2 = v0 * v0 + v1 * v1;
#pragma unroll
    for (int m = 16; m >= 1; m >>= 1) {
        s1 += __shfl_xor(s1, m, 32);
        s2 += __shfl_xor(s2, m, 32);
    }
    if (lane == 0) { rsum[wave] = s1; rsq[wave] = s2; }
    __syncthreads();
    if (t == 0) {
        float a = 0.f, c = 0.f;
#pragma unroll
        for (int i = 0; i < 8; ++i) { a += rsum[i]; c += rsq[i]; }
        rsum[0] = a; rsq[0] = c;
    }
    __syncthreads();
    const float mu  = rsum[0] * (1.f / DIM_);
    const float var = rsq[0] * (1.f / DIM_) - mu * mu;
    const float rs  = rsqrtf(var + 1e-5f);
    const int l = row & (L_ - 1);
    float o0 = (v0 - mu) * rs * w[t]       + b[t];
    float o1 = (v1 - mu) * rs * w[t + 256] + b[t + 256];
    if (hasRel) {
        o0 += rel[(size_t)l * DIM_ + t];
        o1 += rel[(size_t)l * DIM_ + t + 256];
    }
    out[(size_t)row * DIM_ + t]       = (_Float16)o0;
    out[(size_t)row * DIM_ + t + 256] = (_Float16)o1;
}

// ---------------------------------------------------------------------------
// RMSNorm for q (with HD^-0.5 pre-scale) and k, in place on the f16 qkv
// buffer laid out [which][b][h][l][hd].  One wave per 64-wide row.
// ---------------------------------------------------------------------------
__global__ void __launch_bounds__(256)
rms_kernel(_Float16* __restrict__ qkv, const float* __restrict__ gq,
           const float* __restrict__ gk)
{
    const int gid  = blockIdx.x * 8 + (threadIdx.x >> 5);
    const int lane = threadIdx.x & 31;
    const int rows = B_ * H_ * L_;
    const int which = gid / rows;      // 0 = q, 1 = k
    const int rem   = gid % rows;      // (b*H + h)*L + l
    const int h = (rem / L_) % H_;
    _Float16* rowp = qkv + ((size_t)which * rows + rem) * HD_;
    float x0 = (float)rowp[lane], x1 = (float)rowp[lane + 32];
    if (which == 0) { x0 *= 0.125f; x1 *= 0.125f; }   // HD^-0.5
    float ss = x0 * x0 + x1 * x1;
#pragma unroll
    for (int m = 16; m >= 1; m >>= 1) ss += __shfl_xor(ss, m, 32);
    const float den = fmaxf(sqrtf(ss), 1e-12f);
    const float* g  = (which == 0 ? gq : gk) + h * HD_;
    rowp[lane]      = (_Float16)(x0 / den * 8.0f * g[lane]);
    rowp[lane + 32] = (_Float16)(x1 / den * 8.0f * g[lane + 32]);
}

// ---------------------------------------------------------------------------
// Generic WMMA GEMM:  C[M,N] = epilogue(A16[M,K] @ W[N,K]^T + bias)
// Block tile 128x64, 8 waves (4x2), each wave 2x2 16x16 WMMA tiles.
// K sliced by 32; activations staged via GLOBAL_LOAD_ASYNC_TO_LDS_B128
// (ASYNCcnt), weights via sync load + v_cvt f32->f16 + ds_store.
// mode 0: qkv scatter f16  | mode 1: f32 + resid | mode 2: gelu f16
// mode 3: f32 + resid (final output)
// ---------------------------------------------------------------------------
__global__ void __launch_bounds__(256)
gemm_kernel(const _Float16* __restrict__ A, const float* __restrict__ W,
            const float* __restrict__ bias, const float* __restrict__ resid,
            void* __restrict__ outp, int M, int N, int K, int mode)
{
    __shared__ _Float16 sA[128][32];
    __shared__ _Float16 sB[64][32];
    const int tid = threadIdx.x, wave = tid >> 5, lane = tid & 31;
    const int lhal = lane >> 4, l15 = lane & 15;
    const int m0 = blockIdx.y * 128, n0 = blockIdx.x * 64;
    const int wrow = wave & 3, wcol = wave >> 2;

    CF acc[2][2];
#pragma unroll
    for (int ti = 0; ti < 2; ++ti)
#pragma unroll
        for (int tj = 0; tj < 2; ++tj)
#pragma unroll
            for (int r = 0; r < 8; ++r) acc[ti][tj].e[r] = 0.f;

    const int arow = tid >> 1, ach = (tid & 1) * 16;   // 128 rows x 32 halves
    const int brow = tid >> 2, bc8 = (tid & 3) * 8;    // 64 rows x 32 floats
    const uint32_t laBase = (uint32_t)(uintptr_t)(&sA[arow][ach]);

    for (int k0 = 0; k0 < K; k0 += 32) {
        // ---- stage A slice: async memory->LDS, no VGPR data movement
        const uint64_t ga =
            (uint64_t)(uintptr_t)(A + (size_t)(m0 + arow) * K + k0 + ach);
        async_b128(laBase,      ga);
        async_b128(laBase + 16, ga + 16);
        // ---- stage W slice (f32 -> f16 through VGPRs)
        const float* wg = W + (size_t)(n0 + brow) * K + k0 + bc8;
        float4 w0 = *(const float4*)(wg);
        float4 w1 = *(const float4*)(wg + 4);
        sB[brow][bc8 + 0] = (_Float16)w0.x; sB[brow][bc8 + 1] = (_Float16)w0.y;
        sB[brow][bc8 + 2] = (_Float16)w0.z; sB[brow][bc8 + 3] = (_Float16)w0.w;
        sB[brow][bc8 + 4] = (_Float16)w1.x; sB[brow][bc8 + 5] = (_Float16)w1.y;
        sB[brow][bc8 + 6] = (_Float16)w1.z; sB[brow][bc8 + 7] = (_Float16)w1.w;
        if (k0 + 32 < K)  // global_prefetch_b8 of the next A k-slice
            __builtin_prefetch((const void*)(ga + 64), 0, 1);
        wait_async0();
        __syncthreads();

        AF af[2], bf[2];
#pragma unroll
        for (int ti = 0; ti < 2; ++ti) {
            const int r = wrow * 32 + ti * 16 + l15;
#pragma unroll
            for (int j = 0; j < 16; ++j)
                af[ti].e[j] = sA[r][j + 16 * lhal];
        }
#pragma unroll
        for (int tj = 0; tj < 2; ++tj) {
            const int n = wcol * 32 + tj * 16 + l15;
#pragma unroll
            for (int j = 0; j < 16; ++j)
                bf[tj].e[j] = sB[n][j + 16 * lhal];
        }
#pragma unroll
        for (int ti = 0; ti < 2; ++ti)
#pragma unroll
            for (int tj = 0; tj < 2; ++tj)
                acc[ti][tj].v = wmma_f16(af[ti].v, bf[tj].v, acc[ti][tj].v);
        __syncthreads();
    }

    // ---- epilogue
#pragma unroll
    for (int ti = 0; ti < 2; ++ti)
#pragma unroll
        for (int tj = 0; tj < 2; ++tj)
#pragma unroll
            for (int r = 0; r < 8; ++r) {
                const int m = m0 + wrow * 32 + ti * 16 + r + 8 * lhal;
                const int n = n0 + wcol * 32 + tj * 16 + l15;
                float val = acc[ti][tj].e[r] + bias[n];
                if (mode == 0) {            // qkv -> [which][b][h][l][hd] f16
                    const int which = n >> 9, head = (n >> 6) & 7, hd = n & 63;
                    const int bb = m >> 12, ll = m & (L_ - 1);
                    const size_t idx =
                        ((((size_t)which * B_ + bb) * H_ + head) * L_ + ll) * HD_ + hd;
                    ((_Float16*)outp)[idx] = (_Float16)val;
                } else if (mode == 2) {     // exact GELU, f16 out
                    const float g = 0.5f * val * (1.f + erff(val * 0.70710678f));
                    ((_Float16*)outp)[(size_t)m * N + n] = (_Float16)g;
                } else {                    // modes 1,3: f32 + residual
                    ((float*)outp)[(size_t)m * N + n] =
                        val + resid[(size_t)m * N + n];
                }
            }
}

// ---------------------------------------------------------------------------
// Lightning attention: one workgroup (8 waves) per (b,h).
// LDS: kv state 64x64 f32 (16KB), masked scores 128x128 f16 (32KB),
//      V chunk 128x64 f16 (16KB) filled by async memory->LDS copies that
//      overlap the scores phase.
// scores / intra / inter / kv-update all run through v_wmma_f32_16x16x32_f16.
// ---------------------------------------------------------------------------
__global__ void __launch_bounds__(256)
attn_kernel(const _Float16* __restrict__ q, const _Float16* __restrict__ k,
            const _Float16* __restrict__ v, _Float16* __restrict__ o)
{
    __shared__ float    kv[64][64];      // 16 KB
    __shared__ _Float16 sS[128][128];    // 32 KB
    __shared__ _Float16 sV[128][64];     // 16 KB
    const int bh = blockIdx.x, b = bh >> 3, h = bh & 7;
    const int tid = threadIdx.x, wave = tid >> 5, lane = tid & 31;
    const int lhal = lane >> 4, l15 = lane & 15;
    const float s = exp2f(-(float)(h + 1));           // 2^{-8(h+1)/H}, H=8
    const size_t base = (size_t)bh * L_ * HD_;
    const uint32_t lvBase =
        (uint32_t)(uintptr_t)(&sV[0][0]) + (uint32_t)tid * 64;

    for (int i = tid; i < 64 * 64; i += 256) ((float*)kv)[i] = 0.f;
    __syncthreads();

    const int mt   = wave;                // this wave's 16-row M tile
    const int rowc = mt * 16 + l15;       // row within chunk (0..127)

    for (int ci = 0; ci < L_ / CHUNK_; ++ci) {
        const int l0 = ci * CHUNK_;

        // ---- kick off async copy of this chunk's V tile (16 KB) into LDS;
        //      it completes under ASYNCcnt while we compute the scores.
        {
            const uint64_t gv =
                (uint64_t)(uintptr_t)(v + base + (size_t)l0 * HD_) +
                (uint64_t)tid * 64;
            async_b128(lvBase,      gv);
            async_b128(lvBase + 16, gv + 16);
            async_b128(lvBase + 32, gv + 32);
            async_b128(lvBase + 48, gv + 48);
        }

        // A frags of q for this row tile; qd = q * qdec(row) for inter
        AF qa[2], qd[2];
        const float qdec = __expf(-s * (float)rowc);
#pragma unroll
        for (int kf = 0; kf < 2; ++kf)
#pragma unroll
            for (int j = 0; j < 16; ++j) {
                const int kk = kf * 32 + j + 16 * lhal;
                const _Float16 qv = q[base + (size_t)(l0 + rowc) * HD_ + kk];
                qa[kf].e[j] = qv;
                qd[kf].e[j] = (_Float16)((float)qv * qdec);
            }

        // ---- Phase A: scores = (q k^T) .* Dintra -> sS (f16)
        for (int nt = 0; nt < 8; ++nt) {
            AF kb0, kb1;
            const int kcol = nt * 16 + l15;    // key index within chunk
#pragma unroll
            for (int j = 0; j < 16; ++j) {
                const size_t kb = base + (size_t)(l0 + kcol) * HD_;
                kb0.e[j] = k[kb + (j + 16 * lhal)];
                kb1.e[j] = k[kb + 32 + (j + 16 * lhal)];
            }
            CF acc;
#pragma unroll
            for (int r = 0; r < 8; ++r) acc.e[r] = 0.f;
            acc.v = wmma_f16(qa[0].v, kb0.v, acc.v);
            acc.v = wmma_f16(qa[1].v, kb1.v, acc.v);
#pragma unroll
            for (int r = 0; r < 8; ++r) {
                const int i  = mt * 16 + r + 8 * lhal;
                const int jc = nt * 16 + l15;
                const int diff = i - jc;
                const float val =
                    (diff >= 0) ? acc.e[r] * __expf(-s * (float)diff) : 0.f;
                sS[i][jc] = (_Float16)val;
            }
        }
        wait_async0();         // V tile resident in LDS
        __syncthreads();

        // ---- Phase B: out = intra (sS @ V) + inter (q*qdec @ kv)
        for (int nt = 0; nt < 4; ++nt) {
            CF acc;
#pragma unroll
            for (int r = 0; r < 8; ++r) acc.e[r] = 0.f;
            // inter, K over kv rows (64)
#pragma unroll
            for (int kf = 0; kf < 2; ++kf) {
                AF bfr;
#pragma unroll
                for (int j = 0; j < 16; ++j)
                    bfr.e[j] = (_Float16)kv[kf * 32 + j + 16 * lhal][nt * 16 + l15];
                acc.v = wmma_f16(qd[kf].v, bfr.v, acc.v);
            }
            // intra, K over chunk rows (128), V from LDS
#pragma unroll
            for (int kf = 0; kf < 4; ++kf) {
                AF afr, bfr;
#pragma unroll
                for (int j = 0; j < 16; ++j) {
                    const int kk = kf * 32 + j + 16 * lhal;
                    afr.e[j] = sS[rowc][kk];
                    bfr.e[j] = sV[kf * 32 + j + 16 * lhal][nt * 16 + l15];
                }
                acc.v = wmma_f16(afr.v, bfr.v, acc.v);
            }
#pragma unroll
            for (int r = 0; r < 8; ++r) {
                const int i   = mt * 16 + r + 8 * lhal;
                const int col = nt * 16 + l15;
                o[(size_t)(b * L_ + l0 + i) * DIM_ + h * HD_ + col] =
                    (_Float16)acc.e[r];
            }
        }
        __syncthreads();

        // ---- Phase C: kv = kv*cdec + (k*kdec)^T @ V ; 16 tiles over 8 waves
        const float cdec = __expf(-s * (float)CHUNK_);
#pragma unroll
        for (int it = 0; it < 2; ++it) {
            const int t2 = wave * 2 + it;
            const int mt2 = t2 >> 2, nt2 = t2 & 3;
            CF acc;
#pragma unroll
            for (int r = 0; r < 8; ++r) acc.e[r] = 0.f;
#pragma unroll
            for (int kf = 0; kf < 4; ++kf) {
                AF afr, bfr;
#pragma unroll
                for (int j = 0; j < 16; ++j) {
                    const int cc = kf * 32 + j + 16 * lhal;
                    const int dm = mt2 * 16 + l15;
                    afr.e[j] = (_Float16)((float)k[base + (size_t)(l0 + cc) * HD_ + dm]
                                          * __expf(-s * (float)(CHUNK_ - cc)));
                    bfr.e[j] = sV[kf * 32 + j + 16 * lhal][nt2 * 16 + l15];
                }
                acc.v = wmma_f16(afr.v, bfr.v, acc.v);
            }
#pragma unroll
            for (int r = 0; r < 8; ++r) {
                const int d = mt2 * 16 + r + 8 * lhal;
                const int e = nt2 * 16 + l15;
                kv[d][e] = kv[d][e] * cdec + acc.e[r];
            }
        }
        __syncthreads();
    }
}

// ---------------------------------------------------------------------------
// Host-side orchestration
// ---------------------------------------------------------------------------
extern "C" void kernel_launch(void* const* d_in, const int* in_sizes, int n_in,
                              void* d_out, int out_size, void* d_ws, size_t ws_size,
                              hipStream_t stream)
{
    (void)in_sizes; (void)n_in; (void)out_size; (void)ws_size;

    const float* x      = (const float*)d_in[0];
    const float* ln1_w  = (const float*)d_in[1];
    const float* ln1_b  = (const float*)d_in[2];
    const float* rel    = (const float*)d_in[3];
    const float* qkv_w  = (const float*)d_in[4];
    const float* qkv_b  = (const float*)d_in[5];
    const float* gq     = (const float*)d_in[6];
    const float* gk     = (const float*)d_in[7];
    const float* proj_w = (const float*)d_in[8];
    const float* proj_b = (const float*)d_in[9];
    const float* ln2_w  = (const float*)d_in[10];
    const float* ln2_b  = (const float*)d_in[11];
    const float* fc1_w  = (const float*)d_in[12];
    const float* fc1_b  = (const float*)d_in[13];
    const float* fc2_w  = (const float*)d_in[14];
    const float* fc2_b  = (const float*)d_in[15];

    const int BL = B_ * L_;                         // 16384 tokens
    char* ws = (char*)d_ws;
    const size_t smallBytes = (size_t)BL * DIM_ * sizeof(_Float16);     // 16 MB
    const size_t bigBytes   = (size_t)BL * 4 * DIM_ * sizeof(_Float16); // 64 MB
    _Float16* bufSmall = (_Float16*)ws;                 // LN1 -> attnOut -> LN2
    _Float16* bufBig   = (_Float16*)(ws + smallBytes);  // qkv -> ffn1
    float*    x1       = (float*)(ws + smallBytes + bigBytes); // residual f32

    _Float16* q16 = bufBig;
    _Float16* k16 = bufBig + (size_t)BL * DIM_;
    _Float16* v16 = bufBig + (size_t)2 * BL * DIM_;

    // 1. LN1 + rel_bias -> f16
    ln_kernel<<<BL, 256, 0, stream>>>(x, ln1_w, ln1_b, rel, bufSmall, 1);
    // 2. QKV GEMM (16384x512 @ 1536x512^T) -> scattered q/k/v f16
    gemm_kernel<<<dim3(1536 / 64, BL / 128), 256, 0, stream>>>(
        bufSmall, qkv_w, qkv_b, nullptr, bufBig, BL, 1536, DIM_, 0);
    // 3. RMSNorm q,k in place
    rms_kernel<<<(2 * B_ * H_ * L_) / 8, 256, 0, stream>>>(bufBig, gq, gk);
    // 4. Lightning attention -> bufSmall ([token][512] f16)
    attn_kernel<<<B_ * H_, 256, 0, stream>>>(q16, k16, v16, bufSmall);
    // 5. Output projection + residual(x) -> x1 (f32)
    gemm_kernel<<<dim3(DIM_ / 64, BL / 128), 256, 0, stream>>>(
        bufSmall, proj_w, proj_b, x, x1, BL, DIM_, DIM_, 1);
    // 6. LN2 -> f16
    ln_kernel<<<BL, 256, 0, stream>>>(x1, ln2_w, ln2_b, nullptr, bufSmall, 0);
    // 7. FC1 + exact GELU -> f16 (reuses bufBig)
    gemm_kernel<<<dim3(2048 / 64, BL / 128), 256, 0, stream>>>(
        bufSmall, fc1_w, fc1_b, nullptr, bufBig, BL, 2048, DIM_, 2);
    // 8. FC2 + residual(x1) -> d_out (f32)
    gemm_kernel<<<dim3(DIM_ / 64, BL / 128), 256, 0, stream>>>(
        bufBig, fc2_w, fc2_b, x1, (float*)d_out, BL, DIM_, 2048, 3);
}
